// MultiHeadAttention_996432413606
// MI455X (gfx1250) — matrix operand order repaired and checked by reading, not compiled, not run on hardware
//
#include <hip/hip_runtime.h>
#include <hip/hip_bf16.h>

// ---------------------------------------------------------------------------
// MHA for MI455X (gfx1250, wave32). bf16 WMMA, flash-attention, TDM
// (tensor_load_to_lds) staging for GEMM tiles, async global->LDS for K/V,
// double-buffered LDS, softmax row-sum fused into P@V via ones-column.
// ---------------------------------------------------------------------------

#define D_MODEL   1024
#define NUM_HEADS 16
#define HEAD_DIM  64
#define BATCH     2
#define SEQ       2048
#define MTOT      (BATCH * SEQ)

typedef __attribute__((ext_vector_type(16))) __bf16   v16bf;
typedef __attribute__((ext_vector_type(8)))  float    v8f;
typedef __attribute__((ext_vector_type(4)))  unsigned u32x4;
typedef __attribute__((ext_vector_type(8)))  unsigned u32x8;

__device__ __forceinline__ v8f wmma_bf16(v16bf a, v16bf b, v8f c) {
    return __builtin_amdgcn_wmma_f32_16x16x32_bf16(false, a, false, b,
                                                   (short)0, c, false, false);
}

// -------- async global->LDS staging (gfx1250 ASYNCcnt path) ----------------
#if __has_builtin(__builtin_amdgcn_global_load_async_to_lds_b128) && \
    __has_builtin(__builtin_amdgcn_s_wait_asynccnt)
#define USE_ASYNC_LDS 1
#else
#define USE_ASYNC_LDS 0
#endif

#define AS_GLOBAL __attribute__((address_space(1)))
#define AS_LDS    __attribute__((address_space(3)))

typedef int v4i_b128 __attribute__((vector_size(4 * sizeof(int))));

__device__ __forceinline__ void stage_b128(const __bf16* gsrc, __bf16* ldst) {
#if USE_ASYNC_LDS
    __builtin_amdgcn_global_load_async_to_lds_b128(
        (AS_GLOBAL v4i_b128*)gsrc, (AS_LDS v4i_b128*)ldst, 0, 0);
#else
    *(uint4*)ldst = *(const uint4*)gsrc;
#endif
}

__device__ __forceinline__ void stage_wait() {
#if USE_ASYNC_LDS
    __builtin_amdgcn_s_wait_asynccnt(0);
#endif
}

// -------- Tensor Data Mover (TDM) helpers ----------------------------------
__device__ __forceinline__ unsigned lds_byte_offset(const void* p) {
    return (unsigned)(unsigned long long)(AS_LDS const void*)p;
}

// D# for a 2D tile of bf16 (data_size=2B) from a row-major [dim1 x dim0]
// tensor; LDS rows padded 16DW data + 4DW pad (= 40-elem rows, bank-skewed).
// tile_d0 must be 32 elems (64B = 16 DWORDs) for the pad codes below.
__device__ __forceinline__ void tdm_load_2d(const __bf16* gbase, size_t elem_off,
                                            unsigned lds_off,
                                            unsigned tens_d0, unsigned tens_d1,
                                            unsigned tile_d0, unsigned tile_d1) {
    unsigned long long ga = (unsigned long long)gbase + elem_off * 2ull;
    u32x4 g0;
    g0[0] = 1u;                                   // count=1, user desc
    g0[1] = lds_off;                              // lds_addr (bytes)
    g0[2] = (unsigned)ga;                         // global_addr[31:0]
    g0[3] = (unsigned)(ga >> 32) | (2u << 30);    // global_addr[56:32] | type=2
    u32x8 g1;
    g1[0] = (1u << 16)       // data_size = 2B
          | (1u << 20)       // pad_enable
          | (3u << 22)       // pad_interval: 16 DWORDs of data (one 64B row)
          | (3u << 25);      // pad_amount:   4 DWORDs (16B) after each row
    g1[1] = (tens_d0 & 0xFFFFu) << 16;                      // tensor_dim0[15:0]
    g1[2] = (tens_d0 >> 16) | ((tens_d1 & 0xFFFFu) << 16);  // d0[31:16]|d1[15:0]
    g1[3] = (tens_d1 >> 16) | (tile_d0 << 16);              // d1[31:16]|tile_d0
    g1[4] = tile_d1;                                        // tile_dim1 (tile_d2=0)
    g1[5] = tens_d0;                                        // dim0_stride[31:0]
    g1[6] = 0u;                                             // stride[47:32]|d1s lo
    g1[7] = 0u;
    asm volatile("tensor_load_to_lds %0, %1" :: "s"(g0), "s"(g1) : "memory");
}

__device__ __forceinline__ void tdm_wait() {
    __builtin_amdgcn_s_wait_tensorcnt(0);
}

// ---------------------------------------------------------------------------
// f32 -> bf16 conversion (x)
// ---------------------------------------------------------------------------
__global__ void cvt_f32_bf16_kernel(const float* __restrict__ src,
                                    __bf16* __restrict__ dst, int n) {
    int i = blockIdx.x * blockDim.x + threadIdx.x;
    if (i < n) dst[i] = (__bf16)src[i];
}

// ---------------------------------------------------------------------------
// f32 -> bf16 convert + transpose for 1024x1024 weights: dst[n*K+k]=src[k*N+n]
// ---------------------------------------------------------------------------
__global__ __launch_bounds__(256)
void cvt_transpose_kernel(const float* __restrict__ src, __bf16* __restrict__ dst) {
    __shared__ __bf16 tile[32][33];
    const int tx = threadIdx.x, ty = threadIdx.y;
    const int n0 = blockIdx.x * 32, k0 = blockIdx.y * 32;
    #pragma unroll
    for (int i = 0; i < 4; ++i) {
        int k = k0 + ty + i * 8;
        tile[ty + i * 8][tx] = (__bf16)src[(size_t)k * D_MODEL + n0 + tx];
    }
    __syncthreads();
    #pragma unroll
    for (int i = 0; i < 4; ++i) {
        int n = n0 + ty + i * 8;
        dst[(size_t)n * D_MODEL + k0 + tx] = tile[tx][ty + i * 8];
    }
}

// ---------------------------------------------------------------------------
// GEMM: C[4096,1024] = A(bf16 row-major) @ Wt(bf16 PRE-TRANSPOSED [N][K]) + bias
//   MODE 0: bf16 -> [B,H,S,Hd]   MODE 2: bf16 -> [B,H,Hd,S]   MODE 1: f32 out
// 8 waves; block tile 128x64; K-step 32; TDM-staged double-buffered LDS.
// ---------------------------------------------------------------------------
#define GEMM_BM 128
#define GEMM_BN 64
#define GEMM_BK 32
#define LDA_PAD 40       // 32 data elems + 8 pad = TDM pad codes above
#define LDB_PAD 40

template<int MODE>
__global__ __launch_bounds__(256)
void gemm_bf16_kernel(const __bf16* __restrict__ A, const __bf16* __restrict__ Wt,
                      const float* __restrict__ bias,
                      __bf16* __restrict__ dst_bf, float* __restrict__ dst_f32)
{
    constexpr int K = D_MODEL, N = D_MODEL;
    __shared__ __align__(16) __bf16 As [2][GEMM_BM * LDA_PAD];   // [m][k]
    __shared__ __align__(16) __bf16 Bts[2][GEMM_BN * LDB_PAD];   // [n][k]

    const int tid  = threadIdx.x;
    const int lane = tid & 31;
    const int wave = tid >> 5;
    const int wm   = wave >> 1;
    const int wn   = wave & 1;
    const int m0   = blockIdx.y * GEMM_BM;
    const int n0   = blockIdx.x * GEMM_BN;
    const int half = lane >> 4;
    const int l16  = lane & 15;

    // One TDM descriptor pair per k-step moves both tiles (wave 0 issues).
    auto issue_tdm = [&](int buf, int kb) {
        if (wave == 0) {
            tdm_load_2d(A,  (size_t)m0 * K + kb, lds_byte_offset(&As[buf][0]),
                        K, MTOT, GEMM_BK, GEMM_BM);
            tdm_load_2d(Wt, (size_t)n0 * K + kb, lds_byte_offset(&Bts[buf][0]),
                        K, N, GEMM_BK, GEMM_BN);
        }
    };

    v8f c[2][2];
    for (int i = 0; i < 2; ++i)
        for (int j = 0; j < 2; ++j) c[i][j] = (v8f){};

    issue_tdm(0, 0);
    for (int kb = 0, it = 0; kb < K; kb += GEMM_BK, ++it) {
        const int cur = it & 1;
        if (wave == 0) tdm_wait();
        __syncthreads();                        // cur tiles ready, prev reads done
        if (kb + GEMM_BK < K)
            issue_tdm(cur ^ 1, kb + GEMM_BK);   // DMA next while computing

        v16bf af[2], bf[2];
        #pragma unroll
        for (int t = 0; t < 2; ++t) {
            const int row = wm * 32 + t * 16 + l16;
            #pragma unroll
            for (int v = 0; v < 8; ++v) {
                int k0 = half * 16 + 2 * v;
                af[t][2 * v]     = As[cur][row * LDA_PAD + k0];
                af[t][2 * v + 1] = As[cur][row * LDA_PAD + k0 + 1];
            }
            const int col = wn * 32 + t * 16 + l16;
            #pragma unroll
            for (int v = 0; v < 8; ++v) {
                int kk = half * 16 + 2 * v;
                bf[t][2 * v]     = Bts[cur][col * LDB_PAD + kk];
                bf[t][2 * v + 1] = Bts[cur][col * LDB_PAD + kk + 1];
            }
        }
        #pragma unroll
        for (int tm = 0; tm < 2; ++tm)
            #pragma unroll
            for (int tn = 0; tn < 2; ++tn)
                c[tm][tn] = wmma_bf16(af[tm], bf[tn], c[tm][tn]);
    }

    #pragma unroll
    for (int tm = 0; tm < 2; ++tm) {
        #pragma unroll
        for (int tn = 0; tn < 2; ++tn) {
            const int n = n0 + wn * 32 + tn * 16 + l16;
            const float bias_n = bias[n];
            const int mbase = m0 + wm * 32 + tm * 16 + half * 8;
            #pragma unroll
            for (int v = 0; v < 8; ++v) {
                const int m = mbase + v;
                const float val = c[tm][tn][v] + bias_n;
                if (MODE == 0) {          // [B,H,S,Hd]
                    const int bb = m >> 11, s = m & (SEQ - 1);
                    const int h = n >> 6, hd = n & (HEAD_DIM - 1);
                    dst_bf[(((size_t)bb * NUM_HEADS + h) * SEQ + s) * HEAD_DIM + hd] =
                        (__bf16)val;
                } else if (MODE == 2) {   // [B,H,Hd,S] (transposed V)
                    const int bb = m >> 11, s = m & (SEQ - 1);
                    const int h = n >> 6, hd = n & (HEAD_DIM - 1);
                    dst_bf[(((size_t)bb * NUM_HEADS + h) * HEAD_DIM + hd) * SEQ + s] =
                        (__bf16)val;
                } else {                  // f32 row-major
                    dst_f32[(size_t)m * N + n] = val;
                }
            }
        }
    }
}

// ---------------------------------------------------------------------------
// Flash attention: 4 waves x 16 query rows per workgroup for one (b,h).
// 64-key blocks (halves softmax butterfly + rescale cost per key),
// double-buffered async K/V staging, row-sum fused into P@V via ones-column.
// ---------------------------------------------------------------------------
#define FA_KB 64
#define LDK   72
#define LDV   72
#define LDP   72

__global__ __launch_bounds__(128)
void flash_attn_kernel(const __bf16* __restrict__ Qb, const __bf16* __restrict__ Kb,
                       const __bf16* __restrict__ Vtb, __bf16* __restrict__ Ob)
{
    __shared__ __align__(16) __bf16 Ks [2][FA_KB * LDK];      // [key][hd]
    __shared__ __align__(16) __bf16 Vts[2][HEAD_DIM * LDV];   // [hd][key]
    __shared__ __align__(16) __bf16 Ps [4 * 16 * LDP];        // per-wave P 16x64

    const int tid  = threadIdx.x;
    const int lane = tid & 31;
    const int wave = tid >> 5;
    const int half = lane >> 4;
    const int l16  = lane & 15;

    const int bh = blockIdx.y;               // b*H + h
    const int b  = bh >> 4, h = bh & 15;
    const int q0 = blockIdx.x * 64 + wave * 16;

    const __bf16* Qh  = Qb  + (size_t)bh * SEQ * HEAD_DIM;
    const __bf16* Kh  = Kb  + (size_t)bh * SEQ * HEAD_DIM;
    const __bf16* Vth = Vtb + (size_t)bh * HEAD_DIM * SEQ;   // [hd][s]

    auto stage_kv = [&](int buf, int kb) {
        #pragma unroll
        for (int i = 0; i < 4; ++i) {            // K block [64 keys][64 hd]
            int idx = i * 128 + tid;
            int r = idx >> 3, cv = idx & 7;
            stage_b128(Kh + (size_t)(kb + r) * HEAD_DIM + cv * 8,
                       &Ks[buf][r * LDK + cv * 8]);
        }
        #pragma unroll
        for (int i = 0; i < 4; ++i) {            // V block [64 hd][64 keys]
            int idx = i * 128 + tid;
            int r = idx >> 3, cv = idx & 7;
            stage_b128(Vth + (size_t)r * SEQ + kb + cv * 8,
                       &Vts[buf][r * LDV + cv * 8]);
        }
    };

    // Q fragments, pre-scaled by 1/sqrt(Hd) = 0.125 (exact in bf16)
    v16bf aq[2];
    {
        const __bf16* qrow = Qh + (size_t)(q0 + l16) * HEAD_DIM;
        #pragma unroll
        for (int kk = 0; kk < 2; ++kk)
            #pragma unroll
            for (int v = 0; v < 8; ++v) {
                int k0 = kk * 32 + half * 16 + 2 * v;
                aq[kk][2 * v]     = (__bf16)((float)qrow[k0]     * 0.125f);
                aq[kk][2 * v + 1] = (__bf16)((float)qrow[k0 + 1] * 0.125f);
            }
    }

    // ones-column B fragment: column 0 of the extra n-tile is all-1
    v16bf bones;
    #pragma unroll
    for (int j = 0; j < 16; ++j)
        bones[j] = (l16 == 0) ? (__bf16)1.0f : (__bf16)0.0f;

    v8f acc[4];                 // O accumulators (4 head-dim tiles)
    v8f accl = (v8f){};         // row-sum accumulator (ones-column tile)
    for (int i = 0; i < 4; ++i) acc[i] = (v8f){};
    float mrow[8];
    #pragma unroll
    for (int v = 0; v < 8; ++v) mrow[v] = -3.0e38f;

    __bf16* Pw = &Ps[wave * 16 * LDP];

    stage_kv(0, 0);
    for (int kb = 0, it = 0; kb < SEQ; kb += FA_KB, ++it) {
        const int cur = it & 1;
        stage_wait();
        __syncthreads();                  // cur K/V ready; prev reads done
        if (kb + FA_KB < SEQ)
            stage_kv(cur ^ 1, kb + FA_KB);    // overlap DMA with compute

        // scores: four 16-key n-tiles, Hd=64 -> 2 k-steps each
        v8f s[4];
        #pragma unroll
        for (int nt = 0; nt < 4; ++nt) s[nt] = (v8f){};
        #pragma unroll
        for (int nt = 0; nt < 4; ++nt) {
            const int key = nt * 16 + l16;
            #pragma unroll
            for (int kk = 0; kk < 2; ++kk) {
                v16bf bk;
                #pragma unroll
                for (int v = 0; v < 8; ++v) {
                    int hd = kk * 32 + half * 16 + 2 * v;   // contiguous 32B
                    bk[2 * v]     = Ks[cur][key * LDK + hd];
                    bk[2 * v + 1] = Ks[cur][key * LDK + hd + 1];
                }
                s[nt] = wmma_bf16(aq[kk], bk, s[nt]);
            }
        }

        // online softmax over 64 keys: one butterfly per row per block
        #pragma unroll
        for (int v = 0; v < 8; ++v) {
            float t = fmaxf(fmaxf(s[0][v], s[1][v]), fmaxf(s[2][v], s[3][v]));
            t = fmaxf(t, __shfl_xor(t, 1, 32));
            t = fmaxf(t, __shfl_xor(t, 2, 32));
            t = fmaxf(t, __shfl_xor(t, 4, 32));
            t = fmaxf(t, __shfl_xor(t, 8, 32));
            const float mnew = fmaxf(mrow[v], t);
            const float corr = __expf(mrow[v] - mnew);
            mrow[v] = mnew;
            #pragma unroll
            for (int t4 = 0; t4 < 4; ++t4) acc[t4][v] *= corr;
            accl[v] *= corr;
            const int m = half * 8 + v;
            #pragma unroll
            for (int nt = 0; nt < 4; ++nt)
                Pw[m * LDP + nt * 16 + l16] = (__bf16)__expf(s[nt][v] - mnew);
        }
        __syncthreads();   // publish P across lanes

        // [O | l] += P @ [V | 1], K-dim = 64 keys -> 2 k-steps
        v16bf ap[2];
        {
            const __bf16* pr = &Ps[wave * 16 * LDP + l16 * LDP];
            #pragma unroll
            for (int kk = 0; kk < 2; ++kk)
                #pragma unroll
                for (int v = 0; v < 8; ++v) {
                    int k0 = kk * 32 + half * 16 + 2 * v;
                    ap[kk][2 * v]     = pr[k0];
                    ap[kk][2 * v + 1] = pr[k0 + 1];
                }
        }
        #pragma unroll
        for (int nt = 0; nt < 4; ++nt) {
            const int hd = nt * 16 + l16;
            #pragma unroll
            for (int kk = 0; kk < 2; ++kk) {
                v16bf bv;
                #pragma unroll
                for (int v = 0; v < 8; ++v) {
                    int key = kk * 32 + half * 16 + 2 * v;   // contiguous 32B
                    bv[2 * v]     = Vts[cur][hd * LDV + key];
                    bv[2 * v + 1] = Vts[cur][hd * LDV + key + 1];
                }
                acc[nt] = wmma_bf16(ap[kk], bv, acc[nt]);
            }
        }
        accl = wmma_bf16(ap[0], bones, accl);
        accl = wmma_bf16(ap[1], bones, accl);
    }

    // l for row m lives in column 0 -> lane (half*16); broadcast + normalize
    float inv[8];
    #pragma unroll
    for (int v = 0; v < 8; ++v) {
        float l = __shfl(accl[v], lane & 16, 32);
        inv[v] = 1.0f / l;
    }
    #pragma unroll
    for (int nt = 0; nt < 4; ++nt) {
        const int col = h * HEAD_DIM + nt * 16 + l16;
        #pragma unroll
        for (int v = 0; v < 8; ++v) {
            const int s = q0 + half * 8 + v;
            Ob[((size_t)b * SEQ + s) * D_MODEL + col] = (__bf16)(acc[nt][v] * inv[v]);
        }
    }
}

// ---------------------------------------------------------------------------
extern "C" void kernel_launch(void* const* d_in, const int* in_sizes, int n_in,
                              void* d_out, int out_size, void* d_ws, size_t ws_size,
                              hipStream_t stream)
{
    (void)in_sizes; (void)n_in; (void)out_size; (void)ws_size;
    const float* x    = (const float*)d_in[0];
    const float* Wq   = (const float*)d_in[1];
    const float* bq   = (const float*)d_in[2];
    const float* Wk   = (const float*)d_in[3];
    const float* bk   = (const float*)d_in[4];
    const float* Wv   = (const float*)d_in[5];
    const float* bv   = (const float*)d_in[6];
    const float* Wo   = (const float*)d_in[7];
    const float* bo   = (const float*)d_in[8];

    char* ws = (char*)d_ws;
    const size_t MB = 1024u * 1024u;
    __bf16* xbf  = (__bf16*)(ws + 0 * MB);
    __bf16* wqb  = (__bf16*)(ws + 8 * MB);
    __bf16* wkb  = (__bf16*)(ws + 10 * MB);
    __bf16* wvb  = (__bf16*)(ws + 12 * MB);
    __bf16* wob  = (__bf16*)(ws + 14 * MB);
    __bf16* Qbuf = (__bf16*)(ws + 16 * MB);   // [B,H,S,Hd]
    __bf16* Kbuf = (__bf16*)(ws + 24 * MB);   // [B,H,S,Hd]
    __bf16* Vbuf = (__bf16*)(ws + 32 * MB);   // [B,H,Hd,S]  (transposed)
    __bf16* Obuf = (__bf16*)(ws + 40 * MB);   // [B,S,D]; total 48MB

    const int NX = MTOT * D_MODEL;
    cvt_f32_bf16_kernel<<<(NX + 255) / 256, 256, 0, stream>>>(x, xbf, NX);

    dim3 tg(D_MODEL / 32, D_MODEL / 32), tb(32, 8);
    cvt_transpose_kernel<<<tg, tb, 0, stream>>>(Wq, wqb);
    cvt_transpose_kernel<<<tg, tb, 0, stream>>>(Wk, wkb);
    cvt_transpose_kernel<<<tg, tb, 0, stream>>>(Wv, wvb);
    cvt_transpose_kernel<<<tg, tb, 0, stream>>>(Wo, wob);

    dim3 gg(D_MODEL / GEMM_BN, MTOT / GEMM_BM);      // (16, 32)
    gemm_bf16_kernel<0><<<gg, 256, 0, stream>>>(xbf, wqb, bq, Qbuf, nullptr);
    gemm_bf16_kernel<0><<<gg, 256, 0, stream>>>(xbf, wkb, bk, Kbuf, nullptr);
    gemm_bf16_kernel<2><<<gg, 256, 0, stream>>>(xbf, wvb, bv, Vbuf, nullptr);

    dim3 ga(SEQ / 64, BATCH * NUM_HEADS);            // (32, 32)
    flash_attn_kernel<<<ga, 128, 0, stream>>>(Qbuf, Kbuf, Vbuf, Obuf);

    gemm_bf16_kernel<1><<<gg, 256, 0, stream>>>(Obuf, wob, bo, nullptr, (float*)d_out);
}
